// SimpleRNN_n_25443386261715
// MI455X (gfx1250) — matrix operand order repaired and checked by reading, not compiled, not run on hardware
//
#include <hip/hip_runtime.h>

#define T_LEN 512
#define B_SZ  64
#define I_SZ  256
#define H_SZ  512

typedef __attribute__((ext_vector_type(16))) __bf16 v16bf;
typedef __attribute__((ext_vector_type(8)))  float  v8f;

union Frag16 {
  v16bf v;
  unsigned short u[16];
  uint4 q[2];
};

__device__ __forceinline__ unsigned short f32_to_bf16_rne(float f) {
  union { float f; unsigned int u; } x; x.f = f;
  unsigned int r = x.u + 0x7FFFu + ((x.u >> 16) & 1u);
  return (unsigned short)(r >> 16);
}

// ---------------------------------------------------------------------------
// Kernel 1: convert W_ih and W_hh from f32 to bf16 (row-major preserved).
// ---------------------------------------------------------------------------
__global__ void weights_to_bf16(const float* __restrict__ wih,
                                const float* __restrict__ whh,
                                unsigned short* __restrict__ wih_bf,
                                unsigned short* __restrict__ whh_bf) {
  int idx = blockIdx.x * blockDim.x + threadIdx.x;
  const int n_ih = H_SZ * I_SZ;
  if (idx < n_ih) {
    wih_bf[idx] = f32_to_bf16_rne(wih[idx]);
  } else {
    int j = idx - n_ih;              // j < H*H by launch geometry
    whh_bf[j] = f32_to_bf16_rne(whh[j]);
  }
}

// ---------------------------------------------------------------------------
// Kernel 2: xw[t,b,:] = x[t,b,:] @ W_ih^T + bias_ih + bias_hh  -> d_out[t,b,:]
// Flattened GEMM: rows = T*B = 32768 (tile index mt), cols = H = 512.
// Bandwidth-bound (~96 MB @ 23.3 TB/s); WMMA keeps compute free.
// ---------------------------------------------------------------------------
__global__ void __launch_bounds__(256) xw_gemm(
    const float* __restrict__ x,
    const unsigned short* __restrict__ wih_bf,
    const float* __restrict__ bih,
    const float* __restrict__ bhh,
    float* __restrict__ out) {
  const int mt   = blockIdx.x;            // 0..2047
  const int lane = threadIdx.x & 31;
  const int wv   = threadIdx.x >> 5;      // 0..7
  const int n15  = lane & 15;
  const int half = lane >> 4;

  const float* xrow = x + (mt * 16 + n15) * I_SZ;   // A-matrix row for this lane

  v8f acc[4] = {};

  #pragma unroll
  for (int ks = 0; ks < I_SZ / 32; ++ks) {
    // --- A fragment: 16 bf16 per lane, converted from f32 ---
    const int ka = ks * 32 + half * 8;
    float fa[16];
    *(float4*)(fa + 0)  = *(const float4*)(xrow + ka);
    *(float4*)(fa + 4)  = *(const float4*)(xrow + ka + 4);
    *(float4*)(fa + 8)  = *(const float4*)(xrow + ka + 16);
    *(float4*)(fa + 12) = *(const float4*)(xrow + ka + 20);
    Frag16 a;
    #pragma unroll
    for (int i = 0; i < 16; ++i) a.u[i] = f32_to_bf16_rne(fa[i]);

    // --- 4 B fragments + 4 WMMAs ---
    #pragma unroll
    for (int j = 0; j < 4; ++j) {
      const int nt   = wv * 4 + j;
      const int nrow = nt * 16 + n15;               // W_ih row == output column
      const unsigned short* bp = wih_bf + nrow * I_SZ + ks * 32 + half * 8;
      Frag16 b;
      b.q[0] = *(const uint4*)(bp);
      b.q[1] = *(const uint4*)(bp + 16);
      acc[j] = __builtin_amdgcn_wmma_f32_16x16x32_bf16(
          false, a.v, false, b.v, (short)0, acc[j], false, false);
    }
  }

  #pragma unroll
  for (int j = 0; j < 4; ++j) {
    const int nt   = wv * 4 + j;
    const int hcol = nt * 16 + n15;
    const float bsum = bih[hcol] + bhh[hcol];
    #pragma unroll
    for (int r = 0; r < 8; ++r) {
      const int gm = mt * 16 + r + 8 * half;        // global row (t*B + b)
      out[gm * H_SZ + hcol] = acc[j][r] + bsum;
    }
  }
}

// ---------------------------------------------------------------------------
// Kernel 3: sequential scan  h = relu(xw_t + h @ W_hh^T), in-place over d_out.
// 4 blocks (independent batch slices of 16 rows), 512 threads = 16 waves;
// wave wv owns N-tiles {2wv, 2wv+1}.
//   - even-tile W_hh fragments: register-resident for all 512 steps (128 VGPR)
//   - odd-tile  W_hh rows (256 KB bf16): staged ONCE into LDS, read per step
//     with ds_load_b128 (low latency, off the L2 critical path)
//   - h: double-buffered in LDS (32 KB), one barrier per step
//   - xw[t]: loaded into registers BEFORE the k-loop (latency hidden behind
//     32 WMMAs); xw[t+1] prefetched with global_prefetch_b8
// Dynamic LDS: 32 KB hbuf + 256 KB weights = 288 KB (<= 320 KB per WGP;
// only 4 workgroups exist, so single-WG-per-WGP residency is fine).
// ---------------------------------------------------------------------------
__global__ void __launch_bounds__(512, 1) rnn_scan(
    const unsigned short* __restrict__ whh_bf,
    float* __restrict__ out,            // holds xw on entry, h on exit
    float* __restrict__ hfinal) {
  extern __shared__ unsigned short smem[];
  unsigned short* hbuf    = smem;                    // [2][16][H_SZ]
  unsigned short* whh_lds = smem + 2 * 16 * H_SZ;    // [256][H_SZ] odd-tile rows

  const int bslice = blockIdx.x;        // batch slice: rows bslice*16 .. +15
  const int tid  = threadIdx.x;
  const int lane = tid & 31;
  const int wv   = tid >> 5;            // 0..15
  const int n15  = lane & 15;
  const int half = lane >> 4;

  const int nt0 = wv * 2;
  const int nt1 = wv * 2 + 1;

  // h0 = 0
  for (int i = tid; i < 16 * H_SZ; i += 512) hbuf[i] = 0;

  // Stage odd-tile W_hh rows into LDS: local_row lr = (nt1>>1)*16 + r  maps to
  // global row  (lr/16)*32 + 16 + (lr%16).  256 rows x 64 uint4 each.
  {
    const uint4* src = (const uint4*)whh_bf;
    uint4* dst = (uint4*)whh_lds;
    for (int c = tid; c < 256 * 64; c += 512) {
      const int lr = c >> 6;
      const int qi = c & 63;
      const int g  = (lr >> 4) * 32 + 16 + (lr & 15);
      dst[c] = src[g * 64 + qi];
    }
  }

  // Register-cache all B fragments for nt0: 16 ksteps x 8 VGPRs = 128 VGPRs.
  Frag16 bc[16];
  {
    const unsigned short* bp0 = whh_bf + (nt0 * 16 + n15) * H_SZ;
    #pragma unroll
    for (int ks = 0; ks < 16; ++ks) {
      const unsigned short* bp = bp0 + ks * 32 + half * 16;
      bc[ks].q[0] = *(const uint4*)(bp);
      bc[ks].q[1] = *(const uint4*)(bp + 8);
    }
  }
  __syncthreads();

  // Per-lane LDS row base for odd-tile B fragments (local row = wv*16 + n15).
  const unsigned short* b1row = whh_lds + (wv * 16 + n15) * H_SZ;
  const int hcol0 = nt0 * 16 + n15;
  const int hcol1 = nt1 * 16 + n15;

  for (int t = 0; t < T_LEN; ++t) {
    const int cur = t & 1, nxt = cur ^ 1;
    unsigned short* hcur = hbuf + cur * 16 * H_SZ;
    unsigned short* hnxt = hbuf + nxt * 16 * H_SZ;

    float* xwbase = out + (size_t)t * B_SZ * H_SZ + bslice * 16 * H_SZ;

    // Prefetch next step's xw slab (32 KB/block; 512 threads x 64B strides).
    if (t + 1 < T_LEN) {
      const char* pf = (const char*)(xwbase + B_SZ * H_SZ);
      __builtin_prefetch(pf + tid * 64, 0, 3);
    }

    // Hoist this step's xw loads: their latency hides behind the 32 WMMAs.
    float xw0[8], xw1[8];
    #pragma unroll
    for (int r = 0; r < 8; ++r) {
      const int m = r + 8 * half;
      xw0[r] = xwbase[m * H_SZ + hcol0];
      xw1[r] = xwbase[m * H_SZ + hcol1];
    }

    v8f c0 = {}, c1 = {};
    #pragma unroll
    for (int ks = 0; ks < 16; ++ks) {
      // A fragment from LDS h (row = n15; two 16B chunks per lane)
      Frag16 a;
      const unsigned short* ap = hcur + n15 * H_SZ + ks * 32 + half * 16;
      a.q[0] = *(const uint4*)(ap);
      a.q[1] = *(const uint4*)(ap + 8);
      // B fragment for nt1 from LDS weights
      Frag16 b1;
      const unsigned short* bp = b1row + ks * 32 + half * 16;
      b1.q[0] = *(const uint4*)(bp);
      b1.q[1] = *(const uint4*)(bp + 8);

      c0 = __builtin_amdgcn_wmma_f32_16x16x32_bf16(
          false, a.v, false, bc[ks].v, (short)0, c0, false, false);
      c1 = __builtin_amdgcn_wmma_f32_16x16x32_bf16(
          false, a.v, false, b1.v, (short)0, c1, false, false);
    }

    // Epilogue: v = relu(xw + acc). LDS h-stores FIRST (they gate the
    // barrier / next step); f32 global stores can drain past the barrier.
    float v0[8], v1[8];
    #pragma unroll
    for (int r = 0; r < 8; ++r) {
      v0[r] = fmaxf(xw0[r] + c0[r], 0.0f);
      v1[r] = fmaxf(xw1[r] + c1[r], 0.0f);
      const int m = r + 8 * half;
      hnxt[m * H_SZ + hcol0] = f32_to_bf16_rne(v0[r]);
      hnxt[m * H_SZ + hcol1] = f32_to_bf16_rne(v1[r]);
    }
    #pragma unroll
    for (int r = 0; r < 8; ++r) {
      const int m = r + 8 * half;
      xwbase[m * H_SZ + hcol0] = v0[r];
      xwbase[m * H_SZ + hcol1] = v1[r];
      if (t == T_LEN - 1) {
        hfinal[(bslice * 16 + m) * H_SZ + hcol0] = v0[r];
        hfinal[(bslice * 16 + m) * H_SZ + hcol1] = v1[r];
      }
    }
    __syncthreads();
  }
}

// ---------------------------------------------------------------------------
extern "C" void kernel_launch(void* const* d_in, const int* in_sizes, int n_in,
                              void* d_out, int out_size, void* d_ws, size_t ws_size,
                              hipStream_t stream) {
  const float* x    = (const float*)d_in[0];   // [T,B,I]
  const float* wih  = (const float*)d_in[1];   // [H,I]
  const float* whh  = (const float*)d_in[2];   // [H,H]
  const float* bih  = (const float*)d_in[3];   // [H]
  const float* bhh  = (const float*)d_in[4];   // [H]

  float* out    = (float*)d_out;               // [T,B,H] then [B,H]
  float* hfinal = out + (size_t)T_LEN * B_SZ * H_SZ;

  unsigned short* wih_bf = (unsigned short*)d_ws;            // H*I bf16
  unsigned short* whh_bf = wih_bf + H_SZ * I_SZ;             // H*H bf16

  // Allow >64KB dynamic LDS for rnn_scan (host-side attr set; not a stream op,
  // safe under graph capture; idempotent and deterministic).
  const size_t lds_bytes = (size_t)(2 * 16 * H_SZ + 256 * H_SZ) * sizeof(unsigned short);
  (void)hipFuncSetAttribute((const void*)rnn_scan,
                            hipFuncAttributeMaxDynamicSharedMemorySize,
                            (int)lds_bytes);

  // 1) weight conversion: (H*I + H*H) = 393216 elements
  weights_to_bf16<<<(H_SZ * I_SZ + H_SZ * H_SZ) / 256, 256, 0, stream>>>(
      wih, whh, wih_bf, whh_bf);

  // 2) input GEMM over all timesteps: 2048 M-tiles
  xw_gemm<<<(T_LEN * B_SZ) / 16, 256, 0, stream>>>(x, wih_bf, bih, bhh, out);

  // 3) recurrent scan: 4 independent batch slices
  rnn_scan<<<B_SZ / 16, 512, lds_bytes, stream>>>(whh_bf, out, hfinal);
}
